// QuerySATModel_40183714022105
// MI455X (gfx1250) — matrix operand order repaired and checked by reading, not compiled, not run on hardware
//
#include <hip/hip_runtime.h>
#include <hip/hip_bf16.h>
#include <math.h>

// ---------------------------------------------------------------------------
// CDNA5 (gfx1250) implementation of the QuerySAT forward pass.
// All matmuls run on v_wmma_f32_16x16x32_bf16 (wave32 WMMA).
//  - weights pre-permuted into WMMA B-fragment tiles -> waves load B straight
//    from global (L2-resident), no LDS for B
//  - GEMM-to-GEMM activations stored as bf16 (half the HBM traffic)
//  - A tile double-buffered in LDS, one barrier per k-step
//  - each wave computes two 16x16 col-groups (block tile 16x128)
// ---------------------------------------------------------------------------

typedef __attribute__((ext_vector_type(16))) __bf16 v16bf;
typedef __attribute__((ext_vector_type(4)))  __bf16 v4bf;
typedef __attribute__((ext_vector_type(8)))  float  v8f;

static constexpr size_t NV = 50000;   // N_VARS
static constexpr size_t NC = 210000;  // N_CLAUSES
static constexpr size_t NE = 630000;  // N_EDGES
static constexpr size_t NG = 32;      // N_GRAPHS

// ---- workspace layout (float element offsets; bf16 buffers noted) ---------
static constexpr size_t OFF_VARS   = 0;                       // f32  NV*128
static constexpr size_t OFF_CSTATE = OFF_VARS   + NV*128;     // f32  NC*128
static constexpr size_t OFF_CDATA  = OFF_CSTATE + NC*128;     // f32  NC*192 clause_data
static constexpr size_t OFF_CU     = OFF_CDATA  + NC*192;     // bf16 NC*192 clause_unit
static constexpr size_t OFF_HCM    = OFF_CU     + NC*96;      // bf16 NC*224 h_cm
static constexpr size_t OFF_UNIT   = OFF_HCM    + NC*112;     // bf16 NV*320 v1 / unit
static constexpr size_t OFF_H0     = OFF_UNIT   + NV*160;     // bf16 NV*256
static constexpr size_t OFF_H1     = OFF_H0     + NV*128;     // bf16 NV*256
static constexpr size_t OFF_VARSBF = OFF_H1     + NV*128;     // bf16 NV*128 mirror
static constexpr size_t OFF_QUERY  = OFF_VARSBF + NV*64;      // f32  NV*64 query/logits
static constexpr size_t OFF_VGRAD  = OFF_QUERY  + NV*64;      // f32  NV*64
static constexpr size_t OFF_VLOSS  = OFF_VGRAD  + NV*64;      // f32  2*NV*64 (reused: newvars)
static constexpr size_t OFF_LITDEG = OFF_VLOSS  + 2*NV*64;    // 100032
static constexpr size_t OFF_DW     = OFF_LITDEG + 100032;     // 100032
static constexpr size_t OFF_VDW    = OFF_DW     + 100032;     // 50048
static constexpr size_t OFF_VCNT   = OFF_VDW    + 50048;      // 64
static constexpr size_t OFF_CCNT   = OFF_VCNT   + 64;         // 64
static constexpr size_t OFF_PNSUM  = OFF_CCNT   + 64;         // NG*128
static constexpr size_t OFF_PNVAR  = OFF_PNSUM  + NG*128;     // 64
static constexpr size_t OFF_PGL    = OFF_PNVAR  + 64;         // NG*8
static constexpr size_t OFF_GHARD  = OFF_PGL    + 256;        // NG*8
static constexpr size_t OFF_LOSS   = OFF_GHARD  + 256;        // 64
static constexpr size_t OFF_BIAS   = OFF_LOSS   + 64;         // 1360 padded biases
static constexpr size_t OFF_WB     = OFF_BIAS   + 1360;       // bf16 weights

// ---- small device helpers --------------------------------------------------
__device__ __forceinline__ float softplus_f(float x) {
  return (x > 20.f) ? x : log1pf(__expf(x));
}
__device__ __forceinline__ float sigmoid_f(float x) {
  return 1.f / (1.f + __expf(-x));
}

// ---------------------------------------------------------------------------
__global__ void k_fill(float* __restrict__ p, float v, size_t n) {
  size_t i = (size_t)blockIdx.x * blockDim.x + threadIdx.x;
  if (i < n) p[i] = v;
}

// pack f32 weight K x N -> zero-padded bf16 in WMMA B-fragment tile order:
//   Wp[((kt*ncg + cg)*32 + lane)*16 + j] = W[kt*32 + 16*(lane>>4)+j,
//                                            cg*16 + (lane&15)]
__global__ void k_pack_w(const float* __restrict__ w, int K, int N,
                         __bf16* __restrict__ wp, int Kp, int Np) {
  int i = blockIdx.x * blockDim.x + threadIdx.x;
  if (i >= Kp * Np) return;
  int ncg    = Np >> 4;
  int tile   = i >> 9;
  int within = i & 511;
  int lane   = within >> 4;
  int j      = within & 15;
  int cg     = tile % ncg;
  int kt     = tile / ncg;
  int kk     = ((lane >> 4) << 4) + j;
  int k      = (kt << 5) + kk;
  int n      = (cg << 4) + (lane & 15);
  float v = (k < K && n < N) ? w[k * N + n] : 0.f;
  wp[i] = (__bf16)v;
}

__global__ void k_pack_b(const float* __restrict__ b, int N,
                         float* __restrict__ bp, int Np) {
  int i = blockIdx.x * blockDim.x + threadIdx.x;
  if (i < Np) bp[i] = (i < N) ? b[i] : 0.f;
}

// ---------------------------------------------------------------------------
// WMMA GEMM:  C[M x ldc] = act(A[M x lda](:, 0:Kp) @ W[Kp x Np] + bias)
// block = 128 threads (4 waves), tile 16 rows x 128 cols, k-step 32 (bf16).
// A: bf16 row-major, staged + permuted through double-buffered LDS.
// B: per-wave direct global loads of pre-permuted weight fragments.
// out_bf16: 0 -> f32 row-major, 1 -> bf16 row-major.
// Columns in [Np, ldc) are written as 0 (zero-fills consumer K padding).
// ---------------------------------------------------------------------------
__global__ __launch_bounds__(128) void k_gemm_bf16(
    const __bf16* __restrict__ A, int lda, int Kp,
    const __bf16* __restrict__ W, int Np,
    const float* __restrict__ biasp,
    void* __restrict__ Cout, int ldc, int relu, int out_bf16) {
  __shared__ __bf16 sA[2][512];  // double-buffered 16x32 A fragment

  const int tid  = threadIdx.x;
  const int lane = tid & 31;
  const int wid  = tid >> 5;
  const int row0 = blockIdx.x * 16;
  const int cgA  = blockIdx.y * 8 + wid * 2;  // two adjacent col-groups/wave
  const int cgB  = cgA + 1;
  const int ncg  = Np >> 4;
  const bool activeA = (cgA < ncg);           // wave-uniform
  const bool activeB = (cgB < ncg);

  const int m  = tid >> 3;                    // staging row 0..15
  const int kk = (tid & 7) * 4;               // staging k 0,4,...,28

  auto stageA = [&](int buf, int k0) {
    const __bf16* ap = A + (size_t)(row0 + m) * lda + (k0 + kk);
    v4bf av = *(const v4bf*)ap;               // 8B bf16 load
    if (k0 + 32 < Kp) __builtin_prefetch(ap + 32, 0, 3);  // global_prefetch
    int hi = kk >> 4;
    int l  = m + 16 * hi;
    int j0 = kk & 15;                         // 4 consecutive j within lane slot
    *(v4bf*)(&sA[buf][l * 16 + j0]) = av;     // ds_store_b64
  };

  v8f acc0 = {0.f, 0.f, 0.f, 0.f, 0.f, 0.f, 0.f, 0.f};
  v8f acc1 = acc0;

  stageA(0, 0);
  int buf = 0;
  for (int k0 = 0; k0 < Kp; k0 += 32) {
    __syncthreads();
    v16bf a = *(const v16bf*)(&sA[buf][lane * 16]);
    if (k0 + 32 < Kp) stageA(buf ^ 1, k0 + 32);
    size_t tbase = ((size_t)(k0 >> 5) * ncg) << 9;
    if (activeA) {
      v16bf b = *(const v16bf*)(W + tbase + ((size_t)cgA << 9) + lane * 16);
      acc0 = __builtin_amdgcn_wmma_f32_16x16x32_bf16(false, a, false, b,
                                                     (short)0, acc0, false, false);
    }
    if (activeB) {
      v16bf b = *(const v16bf*)(W + tbase + ((size_t)cgB << 9) + lane * 16);
      acc1 = __builtin_amdgcn_wmma_f32_16x16x32_bf16(false, a, false, b,
                                                     (short)0, acc1, false, false);
    }
    buf ^= 1;
  }

  // ---- store (C/D layout: VGPR v -> row v + 8*(lane>>4), col = lane&15)
  const int rbase = row0 + 8 * (lane >> 4);
  float* Cf = (float*)Cout;
  __bf16* Ch = (__bf16*)Cout;
#pragma unroll
  for (int cc = 0; cc < 2; ++cc) {
    int col = (cgA + cc) * 16 + (lane & 15);
    if (col >= ldc) continue;
    const v8f& acc = cc ? acc1 : acc0;
    float bs = (col < Np) ? biasp[col] : 0.f;
#pragma unroll
    for (int v = 0; v < 8; ++v) {
      float o = acc[v] + bs;
      if (relu) o = fmaxf(o, 0.f);
      if (col >= Np) o = 0.f;
      size_t idx = (size_t)(rbase + v) * ldc + col;
      if (out_bf16) Ch[idx] = (__bf16)o;
      else          Cf[idx] = o;
    }
  }
}

// ---------------------------------------------------------------------------
__global__ void k_degrees(const int* __restrict__ vi, const int* __restrict__ pol,
                          float* __restrict__ litdeg) {
  size_t e = (size_t)blockIdx.x * blockDim.x + threadIdx.x;
  if (e >= NE) return;
  int lit = vi[e] + (pol[e] > 0 ? 0 : (int)NV);
  atomicAdd(&litdeg[lit], 1.f);
}

__global__ void k_degw(const float* __restrict__ litdeg,
                       float* __restrict__ dw, float* __restrict__ vdw) {
  size_t i = (size_t)blockIdx.x * blockDim.x + threadIdx.x;
  if (i < 2 * NV) dw[i] = rsqrtf(fmaxf(litdeg[i], 1.f));
  if (i < NV)     vdw[i] = 4.f * rsqrtf(fmaxf(litdeg[i] + litdeg[i + NV], 1.f));
}

__global__ void k_counts(const int* __restrict__ batch, float* __restrict__ cnt,
                         size_t n) {
  size_t i = (size_t)blockIdx.x * blockDim.x + threadIdx.x;
  if (i < n) atomicAdd(&cnt[batch[i]], 1.f);
}

// v1 = [variables(128) | noise_t(4) | zeros(28)]  bf16, stride 160
__global__ void k_build_v1(const float* __restrict__ vars,
                           const float* __restrict__ noise_t,
                           __bf16* __restrict__ v1) {
  size_t i = (size_t)blockIdx.x * blockDim.x + threadIdx.x;
  if (i >= NV * 160) return;
  size_t r = i / 160; int c = (int)(i % 160);
  float v;
  if (c < 128)      v = vars[r * 128 + c];
  else if (c < 132) v = noise_t[r * 4 + (c - 128)];
  else              v = 0.f;
  v1[i] = (__bf16)v;
}

// clause_unit = [clause_state(128) | 4*exp(-sum softplus(q*sign))(64)]  bf16
__global__ __launch_bounds__(192) void k_clause_unit(
    const float* __restrict__ query, const float* __restrict__ cstate,
    const int* __restrict__ vi, const int* __restrict__ pol,
    __bf16* __restrict__ cunit) {
  size_t c = blockIdx.x;
  int j = threadIdx.x;
  if (j < 128) {
    cunit[c * 192 + j] = (__bf16)cstate[c * 128 + j];
  } else {
    int col = j - 128;
    float s = 0.f;
#pragma unroll
    for (int k = 0; k < 3; ++k) {
      size_t e = 3 * c + k;
      int v = vi[e];
      float sgn = (pol[e] > 0) ? 1.f : -1.f;
      s += softplus_f(query[(size_t)v * 64 + col] * sgn);
    }
    cunit[c * 192 + 128 + col] = (__bf16)(4.f * __expf(-s));
  }
}

// grad of sum(exp(-s)) wrt q; 256 threads = 4 edges x 64 cols per block
__global__ __launch_bounds__(256) void k_vgrad(
    const float* __restrict__ query, const __bf16* __restrict__ cunit,
    const int* __restrict__ vi, const int* __restrict__ pol,
    float* __restrict__ vgrad) {
  size_t e = (size_t)blockIdx.x * 4 + (threadIdx.x >> 6);
  int j = threadIdx.x & 63;
  if (e >= NE) return;
  size_t c = e / 3;
  int v = vi[e];
  float sgn = (pol[e] > 0) ? 1.f : -1.f;
  float cl = (float)cunit[c * 192 + 128 + j] * 0.25f;   // exp(-s)
  float q  = query[(size_t)v * 64 + j];
  float g  = -cl * sigmoid_f(q * sgn) * sgn;
  atomicAdd(&vgrad[(size_t)v * 64 + j], g);
}

// scatter variable_loss_all[clause] into per-literal rows (4 edges / block)
__global__ __launch_bounds__(256) void k_vloss(
    const float* __restrict__ cdata, const int* __restrict__ vi,
    const int* __restrict__ pol, float* __restrict__ vloss) {
  size_t e = (size_t)blockIdx.x * 4 + (threadIdx.x >> 6);
  int j = threadIdx.x & 63;
  if (e >= NE) return;
  size_t c = e / 3;
  int lit = vi[e] + (pol[e] > 0 ? 0 : (int)NV);
  atomicAdd(&vloss[(size_t)lit * 64 + j], cdata[c * 192 + j]);
}

// unit = [vgrad*vdw(64) | variables(128) | vloss_pos*dw(64) | vloss_neg*dw(64)]
__global__ __launch_bounds__(320) void k_build_unit(
    const float* __restrict__ vgrad, const float* __restrict__ vdw,
    const float* __restrict__ vars, const float* __restrict__ vloss,
    const float* __restrict__ dw, __bf16* __restrict__ unit) {
  size_t v = blockIdx.x;
  int j = threadIdx.x;
  float o;
  if (j < 64)       o = vgrad[v * 64 + j] * vdw[v];
  else if (j < 192) o = vars[v * 128 + (j - 64)];
  else if (j < 256) o = vloss[v * 64 + (j - 192)] * dw[v];
  else              o = vloss[(NV + v) * 64 + (j - 256)] * dw[NV + v];
  unit[v * 320 + j] = (__bf16)o;
}

// ---- pair-norm (generic over x[rows x stride](:, coloff : coloff+128)) ----
__global__ __launch_bounds__(128) void k_pn_sum(
    const float* __restrict__ x, int stride, int coloff,
    const int* __restrict__ batch, float* __restrict__ sum) {
  size_t r = blockIdx.x;
  int d = threadIdx.x;
  int g = batch[r];
  atomicAdd(&sum[g * 128 + d], x[r * (size_t)stride + coloff + d]);
}

__global__ void k_pn_mean(float* __restrict__ sum, const float* __restrict__ cnt) {
  int i = blockIdx.x * blockDim.x + threadIdx.x;
  if (i >= (int)(NG * 128)) return;
  float c = fmaxf(cnt[i / 128], 1.f);
  sum[i] /= c;                               // becomes mean
}

__global__ __launch_bounds__(128) void k_pn_var(
    const float* __restrict__ x, int stride, int coloff,
    const int* __restrict__ batch, const float* __restrict__ mean,
    float* __restrict__ varsum) {
  __shared__ float red[128];
  size_t r = blockIdx.x;
  int d = threadIdx.x;
  int g = batch[r];
  float xc = x[r * (size_t)stride + coloff + d] - mean[g * 128 + d];
  red[d] = xc * xc;
  __syncthreads();
#pragma unroll
  for (int s = 64; s > 0; s >>= 1) {
    if (d < s) red[d] += red[d + s];
    __syncthreads();
  }
  if (d == 0) atomicAdd(&varsum[g], red[0]);
}

// dst = (x-mean)*rsqrt(var+eps) + 0.1*old ; optional bf16 mirror
__global__ __launch_bounds__(128) void k_pn_apply(
    const float* __restrict__ x, int stride, int coloff,
    const int* __restrict__ batch, const float* __restrict__ mean,
    const float* __restrict__ varsum, const float* __restrict__ cnt,
    const float* __restrict__ oldv, float* __restrict__ dst,
    __bf16* __restrict__ bfdst, int write_bf) {
  size_t r = blockIdx.x;
  int d = threadIdx.x;
  int g = batch[r];
  float var = varsum[g] / (fmaxf(cnt[g], 1.f) * 128.f);
  float rs = rsqrtf(var + 1e-6f);
  float xc = x[r * (size_t)stride + coloff + d] - mean[g * 128 + d];
  float o = xc * rs + 0.1f * oldv[r * 128 + d];
  dst[r * 128 + d] = o;
  if (write_bf) bfdst[r * 128 + d] = (__bf16)o;
}

// ---- final clause evaluation: soft loss + hard satisfaction ---------------
__global__ __launch_bounds__(256) void k_eval(
    const float* __restrict__ logits, const int* __restrict__ vi,
    const int* __restrict__ pol, const int* __restrict__ cb,
    float* __restrict__ pgl, float* __restrict__ ghard) {
  int t = threadIdx.x;
  size_t c = (size_t)blockIdx.x * 32 + (t >> 3);
  int m = t & 7;
  if (c >= NC) return;
  int g = cb[c];
  float s = 0.f, hard = 0.f;
#pragma unroll
  for (int k = 0; k < 3; ++k) {
    size_t e = 3 * c + k;
    int v = vi[e];
    float sgn = (pol[e] > 0) ? 1.f : -1.f;
    float q = logits[(size_t)v * 16 + m];
    s += softplus_f(q * sgn);
    float lh = (q > 0.f) ? 1.f : 0.f;
    hard += (sgn > 0.f) ? lh : (1.f - lh);
  }
  float cv = __expf(-s);
  float pcl = cv * (-logf(1.f - cv + 1e-8f));
  atomicAdd(&pgl[g * 8 + m], pcl);
  atomicAdd(&ghard[g * 8 + m], fminf(hard, 1.f));
}

__global__ __launch_bounds__(32) void k_step_loss(
    const float* __restrict__ pgl, float* __restrict__ loss_acc) {
  int g = threadIdx.x;
  float v[8];
#pragma unroll
  for (int m = 0; m < 8; ++m)
    v[m] = sqrtf(pgl[g * 8 + m] + 1e-6f) - 1e-3f;
#pragma unroll
  for (int i = 1; i < 8; ++i) {              // insertion sort descending
    float key = v[i]; int j = i - 1;
    while (j >= 0 && v[j] < key) { v[j + 1] = v[j]; --j; }
    v[j + 1] = key;
  }
  float dot = 0.f;
#pragma unroll
  for (int m = 0; m < 8; ++m)
    dot += v[m] * (float)((m + 1) * (m + 1));
  atomicAdd(loss_acc, dot / 204.f);          // sum of squares 1..8 = 204
}

__global__ __launch_bounds__(32) void k_solved(
    const float* __restrict__ ghard, const float* __restrict__ ccnt,
    float* __restrict__ out, int t) {
  int g = threadIdx.x;
  float cn = fmaxf(ccnt[g], 1.f);
  float any = 0.f;
#pragma unroll
  for (int m = 0; m < 8; ++m)
    if (ghard[g * 8 + m] / cn > 1.f - 1e-6f) any = 1.f;
  out[1 + t * 32 + g] = any;
}

__global__ void k_final_loss(const float* __restrict__ loss_acc,
                             float* __restrict__ out) {
  if (threadIdx.x == 0) out[0] = loss_acc[0] * 0.25f;
}

// ---------------------------------------------------------------------------
struct WCfg { int iw, ib, K, N, Kp, Np; size_t woff, boff; };
static const WCfg g_w[9] = {
  { 0,  1, 132,  76, 160,  80,      0,    0},   // vq_w0
  { 2,  3,  76,  64,  96,  64,  12800,   80},   // vq_w1
  { 4,  5, 192, 204, 192, 208,  18944,  144},   // cm_w0
  { 6,  7, 204, 192, 224, 192,  58880,  352},   // cm_w1
  { 8,  9, 320, 230, 320, 240, 101888,  544},   // ug_w0
  {10, 11, 230, 230, 256, 240, 178688,  784},   // ug_w1
  {12, 13, 230, 128, 256, 128, 240128, 1024},   // ug_w2
  {14, 15, 128, 128, 128, 128, 272896, 1152},   // vo_w0
  {16, 17, 128,   8, 128,  16, 289280, 1280},   // vo_w1
};

extern "C" void kernel_launch(void* const* d_in, const int* in_sizes, int n_in,
                              void* d_out, int out_size, void* d_ws, size_t ws_size,
                              hipStream_t stream) {
  (void)in_sizes; (void)n_in; (void)out_size; (void)ws_size;
  float* ws = (float*)d_ws;
  __bf16* wb = (__bf16*)(ws + OFF_WB);
  float* bias = ws + OFF_BIAS;

  __bf16* cu     = (__bf16*)(ws + OFF_CU);
  __bf16* hcm    = (__bf16*)(ws + OFF_HCM);
  __bf16* unitb  = (__bf16*)(ws + OFF_UNIT);
  __bf16* h0     = (__bf16*)(ws + OFF_H0);
  __bf16* h1     = (__bf16*)(ws + OFF_H1);
  __bf16* varsbf = (__bf16*)(ws + OFF_VARSBF);

  const int*   vi    = (const int*)d_in[18];
  const int*   ci    = (const int*)d_in[19]; (void)ci;  // = repeat(arange,3)
  const int*   pol   = (const int*)d_in[20];
  const int*   vb    = (const int*)d_in[21];
  const int*   cb    = (const int*)d_in[22];
  const float* noise = (const float*)d_in[23];
  float* out = (float*)d_out;

  auto zero = [&](size_t off, size_t n) {
    k_fill<<<dim3((unsigned)((n + 255) / 256)), 256, 0, stream>>>(ws + off, 0.f, n);
  };

  // ---- setup ---------------------------------------------------------------
  zero(OFF_LITDEG, 100032);
  zero(OFF_VCNT, 64); zero(OFF_CCNT, 64); zero(OFF_LOSS, 64);
  for (int i = 0; i < 9; ++i) {
    const WCfg& w = g_w[i];
    int nw = w.Kp * w.Np;
    k_pack_w<<<dim3((nw + 255) / 256), 256, 0, stream>>>(
        (const float*)d_in[w.iw], w.K, w.N, wb + w.woff, w.Kp, w.Np);
    k_pack_b<<<dim3((w.Np + 255) / 256), 256, 0, stream>>>(
        (const float*)d_in[w.ib], w.N, bias + w.boff, w.Np);
  }
  k_degrees<<<dim3((unsigned)((NE + 255) / 256)), 256, 0, stream>>>(vi, pol, ws + OFF_LITDEG);
  k_degw<<<dim3((unsigned)((2 * NV + 255) / 256)), 256, 0, stream>>>(
      ws + OFF_LITDEG, ws + OFF_DW, ws + OFF_VDW);
  k_counts<<<dim3((unsigned)((NV + 255) / 256)), 256, 0, stream>>>(vb, ws + OFF_VCNT, NV);
  k_counts<<<dim3((unsigned)((NC + 255) / 256)), 256, 0, stream>>>(cb, ws + OFF_CCNT, NC);
  k_fill<<<dim3((unsigned)((NV * 128 + 255) / 256)), 256, 0, stream>>>(ws + OFF_VARS, 1.f, NV * 128);
  k_fill<<<dim3((unsigned)((NC * 128 + 255) / 256)), 256, 0, stream>>>(ws + OFF_CSTATE, 1.f, NC * 128);

  auto gemm = [&](const __bf16* A, int lda, int Kp, const WCfg& w,
                  void* C, int ldc, int relu, int obf, size_t M) {
    dim3 grid((unsigned)(M / 16), (unsigned)((ldc + 127) / 128));
    k_gemm_bf16<<<grid, 128, 0, stream>>>(A, lda, Kp, wb + w.woff, w.Np,
                                          bias + w.boff, C, ldc, relu, obf);
  };

  for (int t = 0; t < 4; ++t) {
    // v1 = [variables | noise[t] | 0] (bf16); vq MLP -> query (f32)
    k_build_v1<<<dim3((unsigned)((NV * 160 + 255) / 256)), 256, 0, stream>>>(
        ws + OFF_VARS, noise + (size_t)t * NV * 4, unitb);
    gemm(unitb, 160, 160, g_w[0], h0, 96, 1, 1, NV);
    gemm(h0,     96,  96, g_w[1], ws + OFF_QUERY, 64, 0, 0, NV);

    // clause energy + clause_unit (bf16); grad wrt query
    k_clause_unit<<<dim3((unsigned)NC), 192, 0, stream>>>(
        ws + OFF_QUERY, ws + OFF_CSTATE, vi, pol, cu);
    zero(OFF_VGRAD, NV * 64);
    k_vgrad<<<dim3((unsigned)((NE + 3) / 4)), 256, 0, stream>>>(
        ws + OFF_QUERY, cu, vi, pol, ws + OFF_VGRAD);

    // cm MLP: clause_unit -> h_cm (bf16) -> clause_data (f32)
    gemm(cu,  192, 192, g_w[2], hcm, 224, 1, 1, NC);
    gemm(hcm, 224, 224, g_w[3], ws + OFF_CDATA, 192, 0, 0, NC);

    // pair-norm clause_data[:,64:192]; clause_state = pn + 0.1*clause_state
    zero(OFF_PNSUM, NG * 128); zero(OFF_PNVAR, 64);
    k_pn_sum<<<dim3((unsigned)NC), 128, 0, stream>>>(ws + OFF_CDATA, 192, 64, cb, ws + OFF_PNSUM);
    k_pn_mean<<<dim3((NG * 128 + 255) / 256), 256, 0, stream>>>(ws + OFF_PNSUM, ws + OFF_CCNT);
    k_pn_var<<<dim3((unsigned)NC), 128, 0, stream>>>(ws + OFF_CDATA, 192, 64, cb,
                                                     ws + OFF_PNSUM, ws + OFF_PNVAR);
    k_pn_apply<<<dim3((unsigned)NC), 128, 0, stream>>>(
        ws + OFF_CDATA, 192, 64, cb, ws + OFF_PNSUM, ws + OFF_PNVAR,
        ws + OFF_CCNT, ws + OFF_CSTATE, ws + OFF_CSTATE, varsbf, 0);

    // variables_loss scatter, unit build (bf16), ug MLP
    zero(OFF_VLOSS, 2 * NV * 64);
    k_vloss<<<dim3((unsigned)((NE + 3) / 4)), 256, 0, stream>>>(
        ws + OFF_CDATA, vi, pol, ws + OFF_VLOSS);
    k_build_unit<<<dim3((unsigned)NV), 320, 0, stream>>>(
        ws + OFF_VGRAD, ws + OFF_VDW, ws + OFF_VARS, ws + OFF_VLOSS,
        ws + OFF_DW, unitb);
    gemm(unitb, 320, 320, g_w[4], h0, 256, 1, 1, NV);
    gemm(h0,    256, 256, g_w[5], h1, 256, 1, 1, NV);
    gemm(h1,    256, 256, g_w[6], ws + OFF_VLOSS, 128, 0, 0, NV);  // newvars

    // pair-norm new variables; variables = pn + 0.1*variables (+bf16 mirror)
    zero(OFF_PNSUM, NG * 128); zero(OFF_PNVAR, 64);
    k_pn_sum<<<dim3((unsigned)NV), 128, 0, stream>>>(ws + OFF_VLOSS, 128, 0, vb, ws + OFF_PNSUM);
    k_pn_mean<<<dim3((NG * 128 + 255) / 256), 256, 0, stream>>>(ws + OFF_PNSUM, ws + OFF_VCNT);
    k_pn_var<<<dim3((unsigned)NV), 128, 0, stream>>>(ws + OFF_VLOSS, 128, 0, vb,
                                                     ws + OFF_PNSUM, ws + OFF_PNVAR);
    k_pn_apply<<<dim3((unsigned)NV), 128, 0, stream>>>(
        ws + OFF_VLOSS, 128, 0, vb, ws + OFF_PNSUM, ws + OFF_PNVAR,
        ws + OFF_VCNT, ws + OFF_VARS, ws + OFF_VARS, varsbf, 1);

    // vo MLP -> logits (f32, stride 16, cols 0..7 valid)
    gemm(varsbf, 128, 128, g_w[7], h1, 128, 1, 1, NV);
    gemm(h1,     128, 128, g_w[8], ws + OFF_QUERY, 16, 0, 0, NV);

    // per-graph loss + hard satisfaction
    zero(OFF_PGL, 256); zero(OFF_GHARD, 256);
    k_eval<<<dim3((unsigned)((NC + 31) / 32)), 256, 0, stream>>>(
        ws + OFF_QUERY, vi, pol, cb, ws + OFF_PGL, ws + OFF_GHARD);
    k_step_loss<<<dim3(1), 32, 0, stream>>>(ws + OFF_PGL, ws + OFF_LOSS);
    k_solved<<<dim3(1), 32, 0, stream>>>(ws + OFF_GHARD, ws + OFF_CCNT, out, t);
  }

  k_final_loss<<<dim3(1), 32, 0, stream>>>(ws + OFF_LOSS, out);
}
